// GNNEncoder_32478542692805
// MI455X (gfx1250) — compile-verified, matrix operand order checked by reading, not run on hardware
//
#include <hip/hip_runtime.h>

typedef __bf16 bf16_t;
typedef bf16_t v16bf __attribute__((ext_vector_type(16)));
typedef float  v8f   __attribute__((ext_vector_type(8)));

#define LN_EPS 1e-5f

// ---------------------------------------------------------------------------
// float atomic max via signed/unsigned int atomics (works for all sign mixes)
// ---------------------------------------------------------------------------
__device__ __forceinline__ void atomicMaxF(float* addr, float val) {
  if (!(val < 0.0f)) {
    atomicMax((int*)addr, __float_as_int(val));
  } else {
    atomicMin((unsigned int*)addr, (unsigned int)__float_as_int(val));
  }
}

// ---------------------------------------------------------------------------
// WMMA fragment load: 16 bf16 (32B) per lane as two 16B chunks.
// A 16x32 bf16 fragment: lane L<16 holds row L, K in {0..7} u {16..23};
// lane L>=16 holds row L-16, K in {8..15} u {24..31}.  B uses the same
// striping on columns, so we feed it rows of the pre-transposed weight W^T.
// ---------------------------------------------------------------------------
__device__ __forceinline__ v16bf load_frag(const bf16_t* p) {
  union { v16bf v; uint4 q[2]; } u;
  u.q[0] = *(const uint4*)(p);
  u.q[1] = *(const uint4*)(p + 16);
  return u.v;
}

// C = A[M,K](bf16) * W[K,N](as W^T bf16) + bias, fp32 out.
// Block = 128 threads = 4 waves; each wave owns a 16x64 output tile.
__global__ void wmma_gemm_bias(const bf16_t* __restrict__ A,
                               const bf16_t* __restrict__ BT,
                               const float* __restrict__ bias,
                               float* __restrict__ C,
                               int M, int K, int Nout) {
  const int lane = threadIdx.x & 31;
  const int wave = threadIdx.x >> 5;
  const int m0 = (blockIdx.x * 4 + wave) * 16;
  const int n0 = blockIdx.y * 64;
  const int hi = lane >> 4;     // 0/1: which K-half this lane carries
  const int r  = lane & 15;

  int arow = m0 + r; if (arow > M - 1) arow = M - 1;   // clamp tail rows
  const bf16_t* ap = A  + (size_t)arow * K + hi * 8;
  const bf16_t* bp = BT + (size_t)(n0 + r) * K + hi * 8;
  const size_t bstep = (size_t)16 * K;

  v8f acc0 = {}; v8f acc1 = {}; v8f acc2 = {}; v8f acc3 = {};
  for (int k = 0; k < K; k += 32) {
    v16bf a  = load_frag(ap + k);
    v16bf f0 = load_frag(bp + k);
    v16bf f1 = load_frag(bp + bstep + k);
    v16bf f2 = load_frag(bp + 2 * bstep + k);
    v16bf f3 = load_frag(bp + 3 * bstep + k);
    acc0 = __builtin_amdgcn_wmma_f32_16x16x32_bf16(false, a, false, f0, (short)0, acc0, false, false);
    acc1 = __builtin_amdgcn_wmma_f32_16x16x32_bf16(false, a, false, f1, (short)0, acc1, false, false);
    acc2 = __builtin_amdgcn_wmma_f32_16x16x32_bf16(false, a, false, f2, (short)0, acc2, false, false);
    acc3 = __builtin_amdgcn_wmma_f32_16x16x32_bf16(false, a, false, f3, (short)0, acc3, false, false);
  }

  // C/D layout: VGPR v, lanes 0-15 -> (M = v, N = lane); lanes 16-31 -> (M = v+8)
  const int mb = m0 + hi * 8;
#pragma unroll
  for (int j = 0; j < 4; ++j) {
    v8f acc = (j == 0) ? acc0 : (j == 1) ? acc1 : (j == 2) ? acc2 : acc3;
    int col = n0 + j * 16 + r;
    float bv = bias[col];
#pragma unroll
    for (int v = 0; v < 8; ++v) {
      int row = mb + v;
      if (row < M) C[(size_t)row * Nout + col] = acc[v] + bv;
    }
  }
}

// ---------------------------------------------------------------------------
// Prep kernels
// ---------------------------------------------------------------------------
__global__ void pack_wT_kernel(const float* __restrict__ W, bf16_t* __restrict__ WT,
                               int K, int Nout) {
  int t = blockIdx.x * blockDim.x + threadIdx.x;
  if (t >= K * Nout) return;
  int k = t / Nout, n = t - k * Nout;
  WT[(size_t)n * K + k] = (bf16_t)W[t];
}

__global__ void f32_to_bf16_kernel(const float* __restrict__ X, bf16_t* __restrict__ Y, long n) {
  long t = (long)blockIdx.x * blockDim.x + threadIdx.x;
  if (t >= n) return;
  Y[t] = (bf16_t)X[t];
}

__global__ void init_layer_kernel(float* __restrict__ mx, float* __restrict__ denom,
                                  float* __restrict__ hbuf, const float* __restrict__ bias,
                                  int N, int heads, int hidden) {
  long t = (long)blockIdx.x * blockDim.x + threadIdx.x;
  long total = (long)N * hidden;
  if (t >= total) return;
  int node = (int)(t / hidden);
  int c = (int)(t - (long)node * hidden);
  hbuf[t] = bias[c];
  if (c < heads) {
    mx[(size_t)node * heads + c] = -3.0e38f;
    denom[(size_t)node * heads + c] = 0.0f;
  }
}

// ---------------------------------------------------------------------------
// Edge phase. A 16-lane group cooperates on one (edge, head): each lane
// issues float4 (b128) loads at consecutive 16B offsets -> coalesced 64-128B
// L2 requests per message; logit reduced with shfl_xor (masks 8..1 stay
// inside the 16-lane half of a wave32).
// ---------------------------------------------------------------------------
__global__ void edge_logits_kernel(const float* __restrict__ xl, const float* __restrict__ xr,
                                   const float* __restrict__ ea, const float* __restrict__ We,
                                   const float* __restrict__ att,
                                   const int* __restrict__ src, const int* __restrict__ dst,
                                   float* __restrict__ logits, float* __restrict__ mx,
                                   int E, int heads, int ch, int hidden) {
  int gid = (blockIdx.x * blockDim.x + threadIdx.x) >> 4;  // (edge, head) index
  int sub = threadIdx.x & 15;
  if (gid >= E * heads) return;
  int e = gid / heads, h = gid - e * heads;
  int s = src[e], d = dst[e];
  float eav = ea[e];
  const float4* pl = (const float4*)(xl + (size_t)s * hidden + h * ch);
  const float4* pr = (const float4*)(xr + (size_t)d * hidden + h * ch);
  const float4* pw = (const float4*)(We + h * ch);
  const float4* pa = (const float4*)(att + h * ch);
  int nq = ch >> 2;                       // 16 (ch=64) or 32 (ch=128)
  float acc = 0.0f;
  for (int q = sub; q < nq; q += 16) {
    float4 l4 = pl[q], r4 = pr[q], w4 = pw[q], a4 = pa[q];
    float m;
    m = l4.x + r4.x + eav * w4.x; m = (m > 0.0f) ? m : 0.2f * m; acc += m * a4.x;
    m = l4.y + r4.y + eav * w4.y; m = (m > 0.0f) ? m : 0.2f * m; acc += m * a4.y;
    m = l4.z + r4.z + eav * w4.z; m = (m > 0.0f) ? m : 0.2f * m; acc += m * a4.z;
    m = l4.w + r4.w + eav * w4.w; m = (m > 0.0f) ? m : 0.2f * m; acc += m * a4.w;
  }
  for (int msk = 8; msk; msk >>= 1) acc += __shfl_xor(acc, msk, 32);
  if (sub == 0) {
    logits[gid] = acc;
    atomicMaxF(&mx[(size_t)d * heads + h], acc);
  }
}

__global__ void edge_exp_kernel(float* __restrict__ logits, const float* __restrict__ mx,
                                float* __restrict__ denom, const int* __restrict__ dst,
                                int E, int heads) {
  int t = blockIdx.x * blockDim.x + threadIdx.x;
  if (t >= E * heads) return;
  int e = t / heads, h = t - e * heads;
  int d = dst[e];
  float a = expf(logits[t] - mx[(size_t)d * heads + h]);
  logits[t] = a;
  atomicAdd(&denom[(size_t)d * heads + h], a);
}

__global__ void edge_agg_kernel(const float* __restrict__ xl, const float* __restrict__ a,
                                const float* __restrict__ denom,
                                const int* __restrict__ src, const int* __restrict__ dst,
                                float* __restrict__ hbuf,
                                int E, int heads, int ch, int hidden) {
  int gid = (blockIdx.x * blockDim.x + threadIdx.x) >> 4;
  int sub = threadIdx.x & 15;
  if (gid >= E * heads) return;
  int e = gid / heads, h = gid - e * heads;
  int s = src[e], d = dst[e];
  float alpha = a[gid] / (denom[(size_t)d * heads + h] + 1e-16f);
  const float4* p = (const float4*)(xl + (size_t)s * hidden + h * ch);
  float* o = hbuf + (size_t)d * hidden + h * ch;
  int nq = ch >> 2;
  for (int q = sub; q < nq; q += 16) {
    float4 v = p[q];
    int c = q << 2;
    atomicAdd(&o[c + 0], v.x * alpha);
    atomicAdd(&o[c + 1], v.y * alpha);
    atomicAdd(&o[c + 2], v.z * alpha);
    atomicAdd(&o[c + 3], v.w * alpha);
  }
}

// ---------------------------------------------------------------------------
// LayerNorm (+ optional ELU). One wave32 per node; float4 loads, shuffle
// reductions, packed 8B bf16 stores for the next layer's WMMA operand.
// ---------------------------------------------------------------------------
__global__ void ln_elu_kernel(const float* __restrict__ hbuf, const float* __restrict__ g,
                              const float* __restrict__ b,
                              float* __restrict__ outf, bf16_t* __restrict__ outbf,
                              int N, int dim, int do_elu) {
  int wave = threadIdx.x >> 5;
  int lane = threadIdx.x & 31;
  int node = blockIdx.x * (blockDim.x >> 5) + wave;
  if (node >= N) return;
  const float4* p4 = (const float4*)(hbuf + (size_t)node * dim);
  const float4* g4 = (const float4*)g;
  const float4* b4 = (const float4*)b;
  int nq = dim >> 7;                     // float4s per lane: 2 (dim=256) or 1
  float4 v4[2];
  float s = 0.0f;
  for (int i = 0; i < nq; ++i) {
    v4[i] = p4[lane + (i << 5)];
    s += v4[i].x + v4[i].y + v4[i].z + v4[i].w;
  }
  for (int m = 16; m; m >>= 1) s += __shfl_xor(s, m, 32);
  float mu = s / (float)dim;
  float var = 0.0f;
  for (int i = 0; i < nq; ++i) {
    float dx = v4[i].x - mu, dy = v4[i].y - mu, dz = v4[i].z - mu, dw = v4[i].w - mu;
    var += dx * dx + dy * dy + dz * dz + dw * dw;
  }
  for (int m = 16; m; m >>= 1) var += __shfl_xor(var, m, 32);
  float rs = rsqrtf(var / (float)dim + LN_EPS);
  for (int i = 0; i < nq; ++i) {
    int q = lane + (i << 5);
    float4 gg = g4[q], bb = b4[q];
    float4 y;
    y.x = (v4[i].x - mu) * rs * gg.x + bb.x;
    y.y = (v4[i].y - mu) * rs * gg.y + bb.y;
    y.z = (v4[i].z - mu) * rs * gg.z + bb.z;
    y.w = (v4[i].w - mu) * rs * gg.w + bb.w;
    if (do_elu) {
      y.x = (y.x > 0.0f) ? y.x : (expf(y.x) - 1.0f);
      y.y = (y.y > 0.0f) ? y.y : (expf(y.y) - 1.0f);
      y.z = (y.z > 0.0f) ? y.z : (expf(y.z) - 1.0f);
      y.w = (y.w > 0.0f) ? y.w : (expf(y.w) - 1.0f);
    }
    if (outf) ((float4*)(outf + (size_t)node * dim))[q] = y;
    if (outbf) {
      union { bf16_t h[4]; uint2 u; } pk;
      pk.h[0] = (bf16_t)y.x; pk.h[1] = (bf16_t)y.y;
      pk.h[2] = (bf16_t)y.z; pk.h[3] = (bf16_t)y.w;
      ((uint2*)(outbf + (size_t)node * dim))[q] = pk.u;
    }
  }
}

// ---------------------------------------------------------------------------
// Host driver
// ---------------------------------------------------------------------------
extern "C" void kernel_launch(void* const* d_in, const int* in_sizes, int n_in,
                              void* d_out, int out_size, void* d_ws, size_t ws_size,
                              hipStream_t stream) {
  const int IN = 128, HID = 256, OUT = 128, H = 4, C1 = HID / H;
  const int N = in_sizes[0] / IN;
  const int E = in_sizes[2];

  const float* x  = (const float*)d_in[0];
  const int* eidx = (const int*)d_in[1];
  const float* ea = (const float*)d_in[2];
  const int* src = eidx;
  const int* dst = eidx + E;

  const float *Wl1=(const float*)d_in[3],  *bl1=(const float*)d_in[4],
              *Wr1=(const float*)d_in[5],  *br1=(const float*)d_in[6],
              *We1=(const float*)d_in[7],  *att1=(const float*)d_in[8],
              *bo1=(const float*)d_in[9],  *g1=(const float*)d_in[10], *be1=(const float*)d_in[11];
  const float *Wl2=(const float*)d_in[12], *bl2=(const float*)d_in[13],
              *Wr2=(const float*)d_in[14], *br2=(const float*)d_in[15],
              *We2=(const float*)d_in[16], *att2=(const float*)d_in[17],
              *bo2=(const float*)d_in[18], *g2=(const float*)d_in[19], *be2=(const float*)d_in[20];
  const float *Wl3=(const float*)d_in[21], *bl3=(const float*)d_in[22],
              *Wr3=(const float*)d_in[23], *br3=(const float*)d_in[24],
              *We3=(const float*)d_in[25], *att3=(const float*)d_in[26],
              *bo3=(const float*)d_in[27], *g3=(const float*)d_in[28], *be3=(const float*)d_in[29];

  // ---- workspace carving (256B aligned) ----
  char* base = (char*)d_ws;
  size_t off = 0;
  auto carve = [&](size_t bytes) -> void* {
    void* p = base + off;
    off += (bytes + 255) & ~(size_t)255;
    return p;
  };
  bf16_t* xbf   = (bf16_t*)carve((size_t)N * HID * sizeof(bf16_t));   // node features bf16
  bf16_t* wlT   = (bf16_t*)carve((size_t)HID * HID * sizeof(bf16_t)); // W^T bf16 (left)
  bf16_t* wrT   = (bf16_t*)carve((size_t)HID * HID * sizeof(bf16_t)); // W^T bf16 (right)
  float*  xl    = (float*)carve((size_t)N * HID * sizeof(float));
  float*  xr    = (float*)carve((size_t)N * HID * sizeof(float));
  float*  hbuf  = (float*)carve((size_t)N * HID * sizeof(float));
  float*  logit = (float*)carve((size_t)E * H * sizeof(float));
  float*  mx    = (float*)carve((size_t)N * H * sizeof(float));
  float*  denom = (float*)carve((size_t)N * H * sizeof(float));
  (void)ws_size; (void)n_in; (void)out_size;

  auto launch_layer = [&](const bf16_t* xin, int K, int heads, int ch,
                          const float* Wl, const float* bl, const float* Wr, const float* br,
                          const float* We, const float* att, const float* bo,
                          const float* g, const float* be,
                          int do_elu, float* outf, bf16_t* outbf) {
    int hidden = heads * ch;
    int wtot = K * hidden;
    pack_wT_kernel<<<(wtot + 255) / 256, 256, 0, stream>>>(Wl, wlT, K, hidden);
    pack_wT_kernel<<<(wtot + 255) / 256, 256, 0, stream>>>(Wr, wrT, K, hidden);

    dim3 gg((N + 63) / 64, hidden / 64);
    wmma_gemm_bias<<<gg, 128, 0, stream>>>(xin, wlT, bl, xl, N, K, hidden);
    wmma_gemm_bias<<<gg, 128, 0, stream>>>(xin, wrT, br, xr, N, K, hidden);

    long itot = (long)N * hidden;
    init_layer_kernel<<<(int)((itot + 255) / 256), 256, 0, stream>>>(mx, denom, hbuf, bo, N, heads, hidden);

    int etot = E * heads;
    long gtot = (long)etot * 16;   // 16 lanes per (edge, head)
    edge_logits_kernel<<<(int)((gtot + 255) / 256), 256, 0, stream>>>(
        xl, xr, ea, We, att, src, dst, logit, mx, E, heads, ch, hidden);
    edge_exp_kernel<<<(etot + 255) / 256, 256, 0, stream>>>(logit, mx, denom, dst, E, heads);
    edge_agg_kernel<<<(int)((gtot + 255) / 256), 256, 0, stream>>>(
        xl, logit, denom, src, dst, hbuf, E, heads, ch, hidden);
    ln_elu_kernel<<<(N + 7) / 8, 256, 0, stream>>>(hbuf, g, be, outf, outbf, N, hidden, do_elu);
  };

  long xtot = (long)N * IN;
  f32_to_bf16_kernel<<<(int)((xtot + 255) / 256), 256, 0, stream>>>(x, xbf, xtot);

  launch_layer(xbf, IN,  H, C1,  Wl1, bl1, Wr1, br1, We1, att1, bo1, g1, be1, 1, nullptr, xbf);
  launch_layer(xbf, HID, H, C1,  Wl2, bl2, Wr2, br2, We2, att2, bo2, g2, be2, 1, nullptr, xbf);
  launch_layer(xbf, HID, 1, OUT, Wl3, bl3, Wr3, br3, We3, att3, bo3, g3, be3, 0, (float*)d_out, nullptr);
}
